// VarRnnCellWrapper_72189810311797
// MI455X (gfx1250) — matrix operand order repaired and checked by reading, not compiled, not run on hardware
//
#include <hip/hip_runtime.h>

#define SEQ   512
#define BATCH 64
#define IDIM  512
#define HDIM  512
#define GDIM  (4 * HDIM)   // 2048
#define NBLK  16           // persistent blocks for the recurrence

typedef __attribute__((ext_vector_type(16))) __bf16 v16bf;
typedef __attribute__((ext_vector_type(8)))  float  v8f;

union ABf16 { v16bf v; uint4 q[2]; };

__device__ __forceinline__ unsigned short f2bf(float f) {
    union { float f; unsigned u; } x; x.f = f;
    unsigned u = x.u;
    u += 0x7fffu + ((u >> 16) & 1u);   // round-to-nearest-even
    return (unsigned short)(u >> 16);
}
__device__ __forceinline__ float fsigmoid(float x) {
    return 1.0f / (1.0f + __expf(-x));
}
__device__ __forceinline__ float ftanh(float x) {
    x = fminf(fmaxf(x, -15.0f), 15.0f);
    float e = __expf(2.0f * x);
    return (e - 1.0f) / (e + 1.0f);
}

// B tile (32x16 bf16, KxN): lane = N col; lanes 0-15 K {0..7,16..23}, lanes 16-31 K {8..15,24..31}
__device__ __forceinline__ void load_b(ABf16& B, const unsigned short* __restrict__ row,
                                       int kb, int lh) {
    B.q[0] = *(const uint4*)(row + kb + lh * 8);
    B.q[1] = *(const uint4*)(row + kb + 16 + lh * 8);
}

// ---- x_bf16[s*B + b][i] = bf16(input * mask_x), packed 2 per u32 ----
__global__ __launch_bounds__(256) void k_mask_convert_x(
    const float* __restrict__ x, const float* __restrict__ mx,
    unsigned* __restrict__ out)
{
    long i = (long)blockIdx.x * blockDim.x + threadIdx.x;
    long n2 = (long)SEQ * BATCH * IDIM / 2;
    if (i >= n2) return;
    long e  = i * 2;
    long me = e % ((long)BATCH * IDIM);
    float a = x[e]     * mx[me];
    float b = x[e + 1] * mx[me + 1];
    out[i] = (unsigned)f2bf(a) | ((unsigned)f2bf(b) << 16);
}

// ---- fp32 -> bf16 convert (weights), packed 2 per u32 ----
__global__ __launch_bounds__(256) void k_convert(
    const float* __restrict__ in, unsigned* __restrict__ out, long n2)
{
    long i = (long)blockIdx.x * blockDim.x + threadIdx.x;
    if (i >= n2) return;
    out[i] = (unsigned)f2bf(in[2 * i]) | ((unsigned)f2bf(in[2 * i + 1]) << 16);
}

// ---- bias[g] = b_ih[g] + b_hh[g] ----
__global__ __launch_bounds__(256) void k_bias(
    const float* __restrict__ bih, const float* __restrict__ bhh,
    float* __restrict__ bias)
{
    int i = blockIdx.x * blockDim.x + threadIdx.x;
    if (i < GDIM) bias[i] = bih[i] + bhh[i];
}

// ---- copy h0/c0 into state; reset the grid barrier counter ----
__global__ __launch_bounds__(256) void k_init(
    const float* __restrict__ h0, const float* __restrict__ c0,
    float* __restrict__ hs, float* __restrict__ cs, unsigned* __restrict__ ctr)
{
    int i = blockIdx.x * blockDim.x + threadIdx.x;
    if (i == 0) *ctr = 0u;
    if (i < BATCH * HDIM) { hs[i] = h0[i]; cs[i] = c0[i]; }
}

// ---- big GEMM: xg[S*B][G] = x_bf16 @ W_ih^T + bias (one 16x64 strip / wave) ----
__global__ __launch_bounds__(256) void k_gemm_xg(
    const unsigned short* __restrict__ xb,   // [S*B][I] bf16 row-major
    const unsigned short* __restrict__ wb,   // [G][I]   bf16 row-major
    const float* __restrict__ bias,          // [G]
    float* __restrict__ xg)                  // [S*B][G] fp32
{
    const int lane = threadIdx.x & 31;
    const int wid  = threadIdx.x >> 5;
    const int gw   = blockIdx.x * 8 + wid;
    const int NT   = GDIM / 64;
    const int m0   = (gw / NT) * 16;
    const int n0   = (gw % NT) * 64;
    const int lm   = lane & 15;
    const int lh   = lane >> 4;

    const unsigned short* arow = xb + (long)(m0 + lm) * IDIM;
    const unsigned short* brow[4];
#pragma unroll
    for (int j = 0; j < 4; ++j)
        brow[j] = wb + (long)(n0 + j * 16 + lm) * IDIM;

    v8f acc[4] = {v8f{}, v8f{}, v8f{}, v8f{}};
    ABf16 A[2], Bm[2][4];

    // A tile (16x32): lanes 0-15 K {0..7,16..23}, lanes 16-31 K {8..15,24..31}
    A[0].q[0] = *(const uint4*)(arow + 0 + lh * 8);
    A[0].q[1] = *(const uint4*)(arow + 16 + lh * 8);
#pragma unroll
    for (int j = 0; j < 4; ++j) load_b(Bm[0][j], brow[j], 0, lh);

    const int NIT = IDIM / 32;               // 16
#pragma unroll
    for (int it = 0; it < NIT; ++it) {
        const int cur = it & 1, nxt = cur ^ 1;
        if (it + 1 < NIT) {
            const int kb = (it + 1) * 32;
            A[nxt].q[0] = *(const uint4*)(arow + kb + lh * 8);
            A[nxt].q[1] = *(const uint4*)(arow + kb + 16 + lh * 8);
#pragma unroll
            for (int j = 0; j < 4; ++j) load_b(Bm[nxt][j], brow[j], kb, lh);
        }
#pragma unroll
        for (int j = 0; j < 4; ++j)
            acc[j] = __builtin_amdgcn_wmma_f32_16x16x32_bf16(
                false, A[cur].v, false, Bm[cur][j].v, (short)0, acc[j], false, false);
    }

#pragma unroll
    for (int j = 0; j < 4; ++j) {
        const int n  = n0 + j * 16 + lm;
        const float bv = bias[n];
#pragma unroll
        for (int r = 0; r < 8; ++r) {
            const int m = m0 + lh * 8 + r;
            xg[(long)m * GDIM + n] = acc[j][r] + bv;
        }
    }
}

// ======================= persistent recurrence kernel =======================
// 16 blocks x 256 threads run ALL 512 timesteps. Per step:
//   1) stage bf16(h * mask_h) into LDS (rotate-swizzled, conflict-free)
//   2) register-prefetch this step's xg slice (completes under the GEMM)
//   3) per-wave 16x16 tile GEMM across all 4 gates (WMMA, B double-buffered)
//   4) LSTM cell update; c lives in REGISTERS across all steps (fixed tile
//      ownership); write h_next (ping-pong) + output[t]
//   5) global-atomic grid barrier (release/acquire) between steps
__global__ __launch_bounds__(256, 1) void k_recurrence(
    const float* __restrict__ xg,            // [S][B][G] (bias pre-added)
    const float* __restrict__ mh,            // [B][H]
    const unsigned short* __restrict__ whh,  // [G][H] bf16 row-major
    float* __restrict__ h_a,                 // ping (holds h0 at entry)
    float* __restrict__ h_b,                 // pong
    float* __restrict__ c_state,             // [B][H] (c0 at entry, c_n at exit)
    float* __restrict__ out,                 // [S][B][H]
    unsigned* __restrict__ ctr)              // grid barrier counter (==0 at entry)
{
    // 64 rows x 512 ushorts, column rotated by 8*row (ushort units):
    // byte addr = row*1024 + ((2*col + 16*row) & 1023)  -> ds_load_b128 conflict-free
    __shared__ unsigned short sH[BATCH * HDIM];   // 64 KB

    const int tid  = threadIdx.x;
    const int lane = tid & 31;
    const int wid  = tid >> 5;                 // 0..7
    const int m0   = (wid & 3) * 16;           // batch tile
    const int n0   = blockIdx.x * 32 + (wid >> 2) * 16;  // hidden tile
    const int lm   = lane & 15;
    const int lh   = lane >> 4;

    const unsigned short* brow[4];
#pragma unroll
    for (int g = 0; g < 4; ++g)
        brow[g] = whh + (long)(g * HDIM + n0 + lm) * HDIM;

    const int arow = m0 + lm;                  // A row index in LDS
    const unsigned short* aBase = sH + arow * HDIM;
    const int arot = arow * 8;                 // rotation (ushort units)

    const int n = n0 + lm;                     // this lane's output column

    // cell state lives in registers for the whole recurrence
    float creg[8];
#pragma unroll
    for (int r = 0; r < 8; ++r)
        creg[r] = c_state[(m0 + lh * 8 + r) * HDIM + n];

    const float* hcur = h_a;
    float* hnxt = h_b;

    for (int t = 0; t < SEQ; ++t) {
        // ---- 1) stage masked h -> LDS (each of 64 rows: 256 threads x 2 cols) ----
        {
            const int col = tid * 2;
            for (int r = 0; r < BATCH; ++r) {
                const float2 hv = *(const float2*)(hcur + r * HDIM + col);
                const float2 mv = *(const float2*)(mh   + r * HDIM + col);
                unsigned pk = (unsigned)f2bf(hv.x * mv.x) |
                              ((unsigned)f2bf(hv.y * mv.y) << 16);
                *(unsigned*)(&sH[r * HDIM + ((col + 8 * r) & (HDIM - 1))]) = pk;
            }
        }
        __syncthreads();

        // ---- 2) prefetch this step's xg slice into registers (HBM latency
        //         overlaps the WMMA loop below) ----
        const float* xg_t = xg + (long)t * BATCH * GDIM;
        float xgv[4][8];
#pragma unroll
        for (int g = 0; g < 4; ++g)
#pragma unroll
            for (int r = 0; r < 8; ++r)
                xgv[g][r] = xg_t[(long)(m0 + lh * 8 + r) * GDIM + g * HDIM + n];

        // ---- 3) GEMM: gates += (h*mask) @ W_hh^T, B double-buffered ----
        v8f acc[4] = {v8f{}, v8f{}, v8f{}, v8f{}};
        ABf16 A[2], Bm[2][4];

        {
            const int i0 = (0 + lh * 8 + arot) & (HDIM - 1);
            const int i1 = (16 + lh * 8 + arot) & (HDIM - 1);
            A[0].q[0] = *(const uint4*)(aBase + i0);
            A[0].q[1] = *(const uint4*)(aBase + i1);
        }
#pragma unroll
        for (int g = 0; g < 4; ++g) load_b(Bm[0][g], brow[g], 0, lh);

        const int NIT = HDIM / 32;             // 16
#pragma unroll
        for (int it = 0; it < NIT; ++it) {
            const int cur = it & 1, nxt = cur ^ 1;
            if (it + 1 < NIT) {
                const int kb = (it + 1) * 32;
                const int i0 = (kb + lh * 8 + arot) & (HDIM - 1);
                const int i1 = (kb + 16 + lh * 8 + arot) & (HDIM - 1);
                A[nxt].q[0] = *(const uint4*)(aBase + i0);
                A[nxt].q[1] = *(const uint4*)(aBase + i1);
#pragma unroll
                for (int g = 0; g < 4; ++g) load_b(Bm[nxt][g], brow[g], kb, lh);
            }
#pragma unroll
            for (int g = 0; g < 4; ++g)
                acc[g] = __builtin_amdgcn_wmma_f32_16x16x32_bf16(
                    false, A[cur].v, false, Bm[cur][g].v, (short)0, acc[g], false, false);
        }

        // ---- 4) LSTM cell update (c in registers) ----
        float* out_t = out + (long)t * BATCH * HDIM;
#pragma unroll
        for (int r = 0; r < 8; ++r) {
            const int m = m0 + lh * 8 + r;
            float gi = acc[0][r] + xgv[0][r];
            float gf = acc[1][r] + xgv[1][r];
            float gg = acc[2][r] + xgv[2][r];
            float go = acc[3][r] + xgv[3][r];
            float cn = fsigmoid(gf) * creg[r] + fsigmoid(gi) * ftanh(gg);
            float hn = fsigmoid(go) * ftanh(cn);
            creg[r] = cn;
            const int hi = m * HDIM + n;
            hnxt[hi]  = hn;
            out_t[hi] = hn;
        }

        // ---- 5) grid barrier (release stores, acquire fresh h) ----
        __syncthreads();                        // also protects sH reuse
        if (t + 1 < SEQ) {
            if (tid == 0) {
                __threadfence();                // device-scope release (writeback)
                __hip_atomic_fetch_add(ctr, 1u, __ATOMIC_RELAXED,
                                       __HIP_MEMORY_SCOPE_AGENT);
                const unsigned want = (unsigned)NBLK * (unsigned)(t + 1);
                while (__hip_atomic_load(ctr, __ATOMIC_RELAXED,
                                         __HIP_MEMORY_SCOPE_AGENT) < want) {
                    __builtin_amdgcn_s_sleep(1);
                }
                __threadfence();                // device-scope acquire (invalidate)
            }
            __syncthreads();
        }

        // swap ping-pong
        const float* tmp = hcur; hcur = hnxt; hnxt = (float*)tmp;
    }

    // write back final cell state (c_n)
#pragma unroll
    for (int r = 0; r < 8; ++r)
        c_state[(m0 + lh * 8 + r) * HDIM + n] = creg[r];
}

// ---- append final h_n, c_n after output (SEQ even -> final h is in h_a) ----
__global__ __launch_bounds__(256) void k_tail(
    const float* __restrict__ hs, const float* __restrict__ cs,
    float* __restrict__ out)
{
    int i = blockIdx.x * blockDim.x + threadIdx.x;
    if (i < BATCH * HDIM) {
        out[(long)SEQ * BATCH * HDIM + i]                = hs[i];
        out[(long)SEQ * BATCH * HDIM + BATCH * HDIM + i] = cs[i];
    }
}

extern "C" void kernel_launch(void* const* d_in, const int* in_sizes, int n_in,
                              void* d_out, int out_size, void* d_ws, size_t ws_size,
                              hipStream_t stream)
{
    const float* x   = (const float*)d_in[0];
    const float* h0  = (const float*)d_in[1];
    const float* c0  = (const float*)d_in[2];
    const float* mx  = (const float*)d_in[3];
    const float* mh  = (const float*)d_in[4];
    const float* Wih = (const float*)d_in[5];
    const float* Whh = (const float*)d_in[6];
    const float* bih = (const float*)d_in[7];
    const float* bhh = (const float*)d_in[8];
    float* out = (float*)d_out;

    // workspace carve-up
    char* p = (char*)d_ws;
    float* xg            = (float*)p;           p += (size_t)SEQ * BATCH * GDIM * 4;
    float* h_a           = (float*)p;           p += (size_t)BATCH * HDIM * 4;
    float* h_b           = (float*)p;           p += (size_t)BATCH * HDIM * 4;
    float* c_state       = (float*)p;           p += (size_t)BATCH * HDIM * 4;
    float* bias          = (float*)p;           p += (size_t)GDIM * 4;
    unsigned* ctr        = (unsigned*)p;        p += 256;   // barrier counter
    unsigned short* xb   = (unsigned short*)p;  p += (size_t)SEQ * BATCH * IDIM * 2;
    unsigned short* wihb = (unsigned short*)p;  p += (size_t)GDIM * IDIM * 2;
    unsigned short* whhb = (unsigned short*)p;  p += (size_t)GDIM * HDIM * 2;
    (void)ws_size; (void)in_sizes; (void)n_in; (void)out_size;

    // 1) convert inputs to bf16 (mask_x applied once, as in reference)
    {
        long n2 = (long)SEQ * BATCH * IDIM / 2;
        k_mask_convert_x<<<(unsigned)((n2 + 255) / 256), 256, 0, stream>>>(x, mx, (unsigned*)xb);
    }
    k_convert<<<(GDIM * IDIM / 2 + 255) / 256, 256, 0, stream>>>(Wih, (unsigned*)wihb, GDIM * IDIM / 2);
    k_convert<<<(GDIM * HDIM / 2 + 255) / 256, 256, 0, stream>>>(Whh, (unsigned*)whhb, GDIM * HDIM / 2);
    k_bias<<<(GDIM + 255) / 256, 256, 0, stream>>>(bih, bhh, bias);
    k_init<<<(BATCH * HDIM + 255) / 256, 256, 0, stream>>>(h0, c0, h_a, c_state, ctr);

    // 2) big parallel GEMM: xg = (x*mask_x) @ W_ih^T + b_ih + b_hh
    k_gemm_xg<<<(SEQ * BATCH / 16) * (GDIM / 64) / 8, 256, 0, stream>>>(xb, wihb, bias, xg);

    // 3) whole recurrence in ONE persistent kernel (16 co-resident blocks)
    k_recurrence<<<NBLK, 256, 0, stream>>>(xg, mh, whhb, h_a, h_b, c_state, out, ctr);

    // 4) append h_n, c_n
    k_tail<<<(BATCH * HDIM + 255) / 256, 256, 0, stream>>>(h_a, c_state, out);
}
